// Int4PackedLinear_67224828117704
// MI455X (gfx1250) — matrix operand order repaired and checked by reading, not compiled, not run on hardware
//
#include <hip/hip_runtime.h>

// CDNA5 / gfx1250: wave32, WMMA f32_16x16x32_f16
typedef _Float16 v16h __attribute__((ext_vector_type(16)));
typedef _Float16 v8h  __attribute__((ext_vector_type(8)));
typedef _Float16 h4   __attribute__((ext_vector_type(4)));
typedef _Float16 h2   __attribute__((ext_vector_type(2)));
typedef float    v8f  __attribute__((ext_vector_type(8)));

#define IN_F    8192
#define OUT_F   8192
#define TOKENS  64
#define PACKED_PER_ROW (IN_F / 8)       // 1024 int32 per output feature
#define NGROUPS        (IN_F / 256)     // 32 quant groups

#define KSPLIT  4
#define KSLICE  (IN_F / KSPLIT)         // 2048 K per ks-group
#define CHUNK   64                      // K staged in LDS per iteration
#define NCHUNK  (KSLICE / CHUNK)        // 32
#define COLS_PER_BLOCK 64               // 4 col-tiles of 16
#define ROWPAD  72                      // fp16 elems per LDS x-row (64 + 8 pad)
#define ASLAB   (TOKENS * ROWPAD)       // fp16 per ks slab (4608)
#define BUFH    (KSPLIT * ASLAB)        // fp16 per ping-pong buffer (18432)
#define SMEM_BYTES (2 * BUFH * 2)       // 73728 B (>= 64KB partial-sum region)

__global__ __launch_bounds__(512)
void int4_linear_wmma_f16(const float* __restrict__ x,
                          const int*   __restrict__ wp,
                          const float* __restrict__ scales,
                          const float* __restrict__ bias,
                          float* __restrict__ out)
{
    __shared__ __align__(16) char smem[SMEM_BYTES];
    _Float16* xs = (_Float16*)smem;

    const int tid  = threadIdx.x;
    const int lane = tid & 31;
    const int wv   = tid >> 5;           // 16 waves
    const int wc   = wv & 3;             // column tile 0..3 (16 cols each)
    const int ks   = wv >> 2;            // K slice 0..3
    const int l16  = lane & 15;
    const int hi   = lane >> 4;          // 0: lanes 0-15, 1: lanes 16-31
    const int tL   = tid & 127;          // thread id within ks-group

    const int colbase = blockIdx.x * COLS_PER_BLOCK;
    const int bcol    = colbase + wc * 16 + l16;    // this lane's weight column

    const int*   __restrict__ wrow = wp     + (size_t)bcol * PACKED_PER_ROW;
    const float* __restrict__ srow = scales + (size_t)bcol * NGROUPS;
    const int ksbase = ks * KSLICE;

    v8f acc[4] = {};     // 4 M-tiles x (16x16 f32)

    // ---- software-pipelined x staging: global fp32 -> regs -> fp16 LDS ----
    float4 regs[8];
    auto loadChunk = [&](int kOff) {
#pragma unroll
        for (int i = 0; i < 8; ++i) {
            const int f4  = tL + 128 * i;     // 0..1023 float4s (64 rows x 16)
            const int row = f4 >> 4;
            const int kq  = f4 & 15;
            regs[i] = *(const float4*)(x + (size_t)row * IN_F + kOff + kq * 4);
        }
    };
    auto storeChunk = [&](int buf) {
#pragma unroll
        for (int i = 0; i < 8; ++i) {
            const int f4  = tL + 128 * i;
            const int row = f4 >> 4;
            const int kq  = f4 & 15;
            h4 h = { (_Float16)regs[i].x, (_Float16)regs[i].y,
                     (_Float16)regs[i].z, (_Float16)regs[i].w };
            *(h4*)(xs + buf * BUFH + ks * ASLAB + row * ROWPAD + kq * 4) = h;
        }
    };

    loadChunk(ksbase);
    storeChunk(0);
    __syncthreads();

    for (int ck = 0; ck < NCHUNK; ++ck) {
        const int kOff = ksbase + ck * CHUNK;
        const int cur  = ck & 1;

        // issue next chunk's global x loads now; latency hides under compute
        if (ck + 1 < NCHUNK) {
            loadChunk(kOff + CHUNK);
            __builtin_prefetch(wrow + ((kOff + 2 * CHUNK) >> 3), 0, 1);
        }

        // dequant constants for this (col, 256-K group)
        const float s = srow[kOff >> 8];
        const _Float16 sh = (_Float16)s;
        const _Float16 ch = (_Float16)(-1032.0f * s);  // (n+1024) offset fold
        const h2 s2 = { sh, sh };
        const h2 c2 = { ch, ch };

        // packed weights for both K-steps of this chunk
        const int2 wb0 = *(const int2*)(wrow + (kOff >> 3) + hi * 2);
        const int2 wb1 = *(const int2*)(wrow + ((kOff + 32) >> 3) + hi * 2);

        const _Float16* abase = xs + cur * BUFH + ks * ASLAB;

#pragma unroll
        for (int j = 0; j < 2; ++j) {        // 2 K-steps of 32 per chunk
            const int2 wbits = j ? wb1 : wb0;

            // ---- B fragment: v_perm_b32 nibble expansion + fp16 magic ----
            const unsigned e0 = ((unsigned)wbits.x) & 0x0F0F0F0Fu;
            const unsigned o0 = (((unsigned)wbits.x) >> 4) & 0x0F0F0F0Fu;
            const unsigned e1 = ((unsigned)wbits.y) & 0x0F0F0F0Fu;
            const unsigned o1 = (((unsigned)wbits.y) >> 4) & 0x0F0F0F0Fu;
            v16h b;
#pragma unroll
            for (int p = 0; p < 4; ++p) {
                // result bytes {0x00, o.byte[p], 0x00, e.byte[p]} -> two fp16
                const unsigned sel = 0x0C000C00u | ((unsigned)p << 16) | (4u + p);
                const unsigned q0 = __builtin_amdgcn_perm(e0, o0, sel) | 0x64006400u;
                const h2 d0 = __builtin_bit_cast(h2, q0) * s2 + c2;  // v_pk_fma_f16
                b[2 * p]     = d0.x;
                b[2 * p + 1] = d0.y;
                const unsigned q1 = __builtin_amdgcn_perm(e1, o1, sel) | 0x64006400u;
                const h2 d1 = __builtin_bit_cast(h2, q1) * s2 + c2;
                b[8 + 2 * p]     = d1.x;
                b[8 + 2 * p + 1] = d1.y;
            }

            // ---- 4 M-tiles: A fragments from LDS, one WMMA each ----
            const int kk = j * 32;
#pragma unroll
            for (int mt = 0; mt < 4; ++mt) {
                // f16 A layout: hi=0 lanes take K kk..+15 contiguous;
                // hi=1 lanes take kk+16..+31 (matches B nibble order).
                const _Float16* ap = abase + (mt * 16 + l16) * ROWPAD + kk + hi * 16;
                const v8h alo = *(const v8h*)ap;           // ds_load_b128
                const v8h ahi = *(const v8h*)(ap + 8);     // ds_load_b128
                const v16h a = __builtin_shufflevector(
                    alo, ahi, 0, 1, 2, 3, 4, 5, 6, 7, 8, 9, 10, 11, 12, 13, 14, 15);
                acc[mt] = __builtin_amdgcn_wmma_f32_16x16x32_f16(
                    false, a, false, b, (short)0, acc[mt], false, false);
            }
        }

        // convert + park next chunk in the other buffer; barrier publishes it
        if (ck + 1 < NCHUNK) {
            storeChunk(1 - cur);
        }
        __syncthreads();
    }

    // ---- deterministic in-LDS K-split reduction + bias + store ----
    float* psum = (float*)smem;          // [ks][m 0..63][c 0..63]
    const int cloc = wc * 16 + l16;
#pragma unroll
    for (int mt = 0; mt < 4; ++mt) {
#pragma unroll
        for (int r = 0; r < 8; ++r) {
            const int m = mt * 16 + hi * 8 + r;   // C/D layout: VGPR r -> M
            psum[((ks * TOKENS + m) << 6) | cloc] = acc[mt][r];
        }
    }
    __syncthreads();

#pragma unroll
    for (int i = 0; i < 8; ++i) {
        const int flat = tid + 512 * i;           // m*64 + c, 0..4095
        const int m = flat >> 6;
        const int c = flat & 63;
        const float v = psum[flat] + psum[4096 + flat] +
                        psum[8192 + flat] + psum[12288 + flat];
        out[(size_t)m * OUT_F + colbase + c] = v + bias[colbase + c];
    }
}

extern "C" void kernel_launch(void* const* d_in, const int* in_sizes, int n_in,
                              void* d_out, int out_size, void* d_ws, size_t ws_size,
                              hipStream_t stream) {
    (void)in_sizes; (void)n_in; (void)out_size; (void)d_ws; (void)ws_size;
    const float* x      = (const float*)d_in[0];
    const int*   wp     = (const int*)  d_in[1];
    const float* scales = (const float*)d_in[2];
    const float* bias   = (const float*)d_in[3];
    float*       out    = (float*)d_out;

    dim3 grid(OUT_F / COLS_PER_BLOCK);   // 128 blocks
    dim3 block(512);                     // 16 waves: 4 col-tiles x 4 K-slices
    int4_linear_wmma_f16<<<grid, block, 0, stream>>>(x, wp, scales, bias, out);
}
